// WrapperTiledV1_32134945309180
// MI455X (gfx1250) — matrix operand order repaired and checked by reading, not compiled, not run on hardware
//
#include <hip/hip_runtime.h>

typedef __attribute__((ext_vector_type(16))) _Float16 v16h;
typedef __attribute__((ext_vector_type(8)))  float    v8f;

#define GH 192
#define GW 192
#define NG 1024
#define NPIX (GH * GW)
#define PIX_PER_BLOCK 128                 // 8 waves * 16 pixels
#define NBLOCKS (NPIX / PIX_PER_BLOCK)    // 288

#if __has_builtin(__builtin_amdgcn_exp2f)
#define EXP2F(x) __builtin_amdgcn_exp2f(x)
#else
#define EXP2F(x) exp2f(x)
#endif

// searchsorted(linspace(0,1,5), v, side='left')  ==  clamp(ceil(4v), 0, 5)
__device__ __forceinline__ int ssl(float v) {
    int i = (int)ceilf(v * 4.0f);
    return min(5, max(0, i));
}

__global__ __launch_bounds__(256) void splat_wmma_kernel(
    const float* __restrict__ x,          // (NPIX, 2)
    const float* __restrict__ mus_logit,  // (NG, 2)
    const float* __restrict__ covs,       // (NG, 2, 2)
    const float* __restrict__ cols,       // (NG, 3)
    float* __restrict__ out)              // (NPIX, 3)
{
    // SoA Gaussian params: broadcast ds_load_b128 / ds_load_b64 per element
    __shared__ float4 s_p4[NG];                       // mux, muy, ca, cb      16 KB
    __shared__ float2 s_p2[NG];                       // cc, tilemask bits      8 KB
    __shared__ __align__(32) _Float16 s_colT[4 * NG]; // colors transposed      8 KB

    const int tid = threadIdx.x;

    // ---- Preprocess 1024 Gaussians cooperatively ----
    for (int g = tid; g < NG; g += 256) {
        const float mux = 1.0f / (1.0f + __expf(-mus_logit[2 * g + 0]));
        const float muy = 1.0f / (1.0f + __expf(-mus_logit[2 * g + 1]));
        const float a  = covs[4 * g + 0];
        const float b  = covs[4 * g + 1];
        const float cc_ = covs[4 * g + 2];
        const float d  = covs[4 * g + 3];
        const float inv = 1.0f / (a * d - b * cc_);
        const float SC = -0.7213475204444817f;        // -0.5 * log2(e)
        const float ca = SC * d * inv;                // * dx*dx
        const float cb = -SC * (b + cc_) * inv;       // * dx*dy
        const float cq = SC * a * inv;                // * dy*dy

        // Tile coverage mask: bit (tiy*4+tix) set iff glo<=ti<ghi in both dims
        const float kb = 3.0348542587702925f;          // sqrt(-2 ln 0.01)
        const float sx = sqrtf(a), sy = sqrtf(d);
        const int glox = ssl(mux - kb * sx) - 1, ghix = ssl(mux + kb * sx);
        const int gloy = ssl(muy - kb * sy) - 1, ghiy = ssl(muy + kb * sy);
        const int lox = max(glox, 0), hix = min(ghix, 4);
        const int loy = max(gloy, 0), hiy = min(ghiy, 4);
        unsigned xm = (lox < hix) ? ((1u << hix) - (1u << lox)) : 0u;
        unsigned tm = 0u;
        for (int t = loy; t < hiy; ++t) tm |= xm << (4 * t);

        s_p4[g] = make_float4(mux, muy, ca, cb);
        s_p2[g] = make_float2(cq, __uint_as_float(tm));

        // Transposed colors; row 3 = zeros. Columns n>=3 of D are never stored,
        // so lanes n in [4,16) may read any row (n&3) without masking.
        s_colT[0 * NG + g] = (_Float16)cols[3 * g + 0];
        s_colT[1 * NG + g] = (_Float16)cols[3 * g + 1];
        s_colT[2 * NG + g] = (_Float16)cols[3 * g + 2];
        s_colT[3 * NG + g] = (_Float16)0.0f;
    }
    __syncthreads();

    // ---- Wave tiling: 16 pixels per wave (A rows), K = Gaussians ----
    const int lane   = tid & 31;
    const int wave   = tid >> 5;
    const int laneHi = lane >> 4;          // K-half selector
    const int n      = lane & 15;          // A row M / output column N
    const int pixBase = blockIdx.x * PIX_PER_BLOCK + wave * 16;
    const int p = pixBase + n;

    const float px = x[2 * p + 0];
    const float py = x[2 * p + 1];
    const int tix = min(3, max(0, ssl(px) - 1));
    const int tiy = min(3, max(0, ssl(py) - 1));
    const int tbit = tiy * 4 + tix;
    const int colRow = (n & 3) * NG;

    // One K=32 chunk: build A (16x32 w tile) branch-free, load B, issue WMMA.
    auto chunk = [&](int g0, v8f acc) -> v8f {
        v16h A;
        #pragma unroll
        for (int h = 0; h < 16; ++h) {
            // contiguous k order used here (B matches): K = h + 16*(lane>=16)
            const int g = g0 + h + (laneHi << 4);
            const float4 P = s_p4[g];
            const float2 Q = s_p2[g];
            const float dx = px - P.x;
            const float dy = py - P.y;
            // s = ca*dx^2 + cb*dx*dy + cq*dy^2  (already scaled by -0.5*log2e)
            const float s = fmaf(dx, fmaf(P.z, dx, P.w * dy), Q.x * dy * dy);
            const float e = EXP2F(s);
            // pure-dataflow mask: multiply by 0.0/1.0 (no select -> no branch)
            const unsigned tm = __float_as_uint(Q.y);
            const float mfl = (float)((tm >> tbit) & 1u);
            A[h] = (_Float16)(e * mfl);
        }
        // B: 16 contiguous halves of transposed color row (2x ds_load_b128)
        const v16h B = *(const v16h*)&s_colT[colRow + g0 + (laneHi << 4)];
        return __builtin_amdgcn_wmma_f32_16x16x32_f16(
            false, A, false, B, (short)0, acc, false, false);
    };

    // Two accumulators break the WMMA->WMMA RAW dependency chain
    v8f acc0 = {0.f, 0.f, 0.f, 0.f, 0.f, 0.f, 0.f, 0.f};
    v8f acc1 = {0.f, 0.f, 0.f, 0.f, 0.f, 0.f, 0.f, 0.f};
    for (int g0 = 0; g0 < NG; g0 += 64) {
        acc0 = chunk(g0, acc0);
        acc1 = chunk(g0 + 32, acc1);
    }

    // C/D layout: element r of lane = (M = r + 8*laneHi, N = lane%16)
    if (n < 3) {
        #pragma unroll
        for (int r = 0; r < 8; ++r) {
            const int m = r + (laneHi << 3);
            out[(pixBase + m) * 3 + n] = acc0[r] + acc1[r];
        }
    }
}

extern "C" void kernel_launch(void* const* d_in, const int* in_sizes, int n_in,
                              void* d_out, int out_size, void* d_ws, size_t ws_size,
                              hipStream_t stream) {
    const float* x         = (const float*)d_in[0];
    const float* mus_logit = (const float*)d_in[1];
    const float* covs      = (const float*)d_in[2];
    const float* cols      = (const float*)d_in[3];
    float* out = (float*)d_out;
    (void)in_sizes; (void)n_in; (void)out_size; (void)d_ws; (void)ws_size;

    splat_wmma_kernel<<<NBLOCKS, 256, 0, stream>>>(x, mus_logit, covs, cols, out);
}
